// InvariantPointAttention_63101659513361
// MI455X (gfx1250) — matrix operand order repaired and checked by reading, not compiled, not run on hardware
//
#include <hip/hip_runtime.h>
#include <hip/hip_bf16.h>
#include <math.h>

// ---------------------------------------------------------------------------
// Invariant Point Attention on MI455X (gfx1250, wave32, WMMA).
// All contractions on v_wmma_f32_16x16x32_bf16 (fp32 accumulate), all operand
// fragments loaded as contiguous b128 runs (weights pre-transposed, V/V_pts
// stored transposed), branchless edge handling, z staged through LDS in the
// HBM-critical o_pair pass, vectorized softmax (b128 LDS scans, packed bf16
// stores).
// ---------------------------------------------------------------------------

typedef __attribute__((ext_vector_type(16))) __bf16 v16bf;
typedef __attribute__((ext_vector_type(8)))  __bf16 v8bf;
typedef __attribute__((ext_vector_type(4)))  __bf16 v4bf;
typedef __attribute__((ext_vector_type(8)))  float  v8f;

#define NRES   768
#define NHEAD  12
#define DPAIR  128
#define DSING  384
#define CATDIM 2112   // 12*(16 + 24 + 8 + 128)

__device__ __forceinline__ v8f wmma_bf16(v16bf a, v16bf b, v8f c) {
  return __builtin_amdgcn_wmma_f32_16x16x32_bf16(false, a, false, b, (short)0, c,
                                                 false, false);
}

__device__ __forceinline__ float fmaxq(float a, float b) { return a > b ? a : b; }

// Load 8 contiguous floats as two b128 loads.
__device__ __forceinline__ void load8f(const float* __restrict__ p, float* d) {
  float4 u0 = *(const float4*)(p);
  float4 u1 = *(const float4*)(p + 4);
  d[0] = u0.x; d[1] = u0.y; d[2] = u0.z; d[3] = u0.w;
  d[4] = u1.x; d[5] = u1.y; d[6] = u1.z; d[7] = u1.w;
}

// ---------------------------------------------------------------------------
// Generic WMMA GEMM with transposed B:  C[M,N] = A[M,K] @ BT[N,K]^T (+bias[N]).
// One wave / 16x16 tile.  Requires M%16==0, K%32==0.  GUARDN handles N%16!=0
// branchlessly (clamped loads + cndmask), used only for the z@Wb bias GEMM.
// ---------------------------------------------------------------------------
template <bool GUARDN>
__global__ void __launch_bounds__(32)
gemm_bt(const float* __restrict__ A, const float* __restrict__ BT,
        const float* __restrict__ biasv, float* __restrict__ C,
        int M, int N, int K, int lda, int ldc) {
  const int ntiles = (N + 15) >> 4;
  const int row0 = (blockIdx.x / ntiles) * 16;
  const int col0 = (blockIdx.x % ntiles) * 16;
  const int lane = threadIdx.x & 31;
  const int nlo = lane & 15, half = lane >> 4;

  const int r = row0 + nlo;
  const int c = col0 + nlo;
  int cc = c;
  bool cok = true;
  if (GUARDN) { cok = (c < N); cc = cok ? c : (N - 1); }

  const float* __restrict__ arow = A + (size_t)r * lda;
  const float* __restrict__ brow = BT + (size_t)cc * K;

  v8f acc = {};
  for (int k0 = 0; k0 < K; k0 += 32) {
    float af[16], bf[16];
    load8f(arow + k0 + half * 8, af);          // A: K-run  [k0+half*8 .. +7]
    load8f(arow + k0 + 16 + half * 8, af + 8); //    K-run  [k0+16+half*8 ..]
    load8f(brow + k0 + half * 8, bf);          // B: K-run  [k0+half*8 .. +7]
    load8f(brow + k0 + 16 + half * 8, bf + 8); //    K-run  [k0+16+half*8 ..]
    v16bf a, b;
#pragma unroll
    for (int e = 0; e < 16; ++e) {
      a[e] = (__bf16)af[e];
      float vb = bf[e];
      if (GUARDN) vb = cok ? vb : 0.f;
      b[e] = (__bf16)vb;
    }
    acc = wmma_bf16(a, b, acc);
  }
  const float bia = biasv ? biasv[cc] : 0.f;
#pragma unroll
  for (int v = 0; v < 8; ++v) {
    int rr = row0 + v + half * 8;
    if (!GUARDN || cok) C[(size_t)rr * ldc + c] = acc[v] + bia;
  }
}

// ---------------------------------------------------------------------------
// Tiny one-off weight transpose:  BT[n,k] = B[k,n].
// ---------------------------------------------------------------------------
__global__ void __launch_bounds__(256)
transpose_kernel(const float* __restrict__ B, float* __restrict__ BT, int K, int N) {
  int idx = blockIdx.x * 256 + threadIdx.x;
  if (idx < K * N) {
    int k = idx / N, n = idx % N;
    BT[(size_t)n * K + k] = B[idx];
  }
}

// ---------------------------------------------------------------------------
// Frames: split kv (k row-major, v transposed), rotate points into the global
// frame (q_pts/k_pts row-major, v_pts transposed), |pt|^2 sums.
// ---------------------------------------------------------------------------
__global__ void __launch_bounds__(192)
frames_kernel(const float* __restrict__ kvraw, const float* __restrict__ qpraw,
              const float* __restrict__ kvpraw, const float* __restrict__ rot,
              const float* __restrict__ trans, float* __restrict__ kbuf,
              float* __restrict__ vT, float* __restrict__ qpts,
              float* __restrict__ kpts, float* __restrict__ vptsT,
              float* __restrict__ qsq, float* __restrict__ ksq) {
  const int n = blockIdx.x;
  const int t = threadIdx.x;  // 192
  {  // kv[..., :16] -> k (row major), kv[..., 16:] -> v (transposed [h*16+d][n])
    int h = t >> 4, d = t & 15;
    kbuf[n * 192 + t] = kvraw[n * 384 + h * 32 + d];
    vT[(size_t)t * NRES + n] = kvraw[n * 384 + h * 32 + 16 + d];
  }
  float R[9], T[3];
#pragma unroll
  for (int i = 0; i < 9; ++i) R[i] = rot[n * 9 + i];
#pragma unroll
  for (int i = 0; i < 3; ++i) T[i] = trans[n * 3 + i];

  if (t < 48) {  // q points: raw = [x(48) | y(48) | z(48)]
    float x = qpraw[n * 144 + t], y = qpraw[n * 144 + 48 + t],
          zc = qpraw[n * 144 + 96 + t];
#pragma unroll
    for (int c = 0; c < 3; ++c)
      qpts[n * 144 + t * 3 + c] =
          R[c * 3 + 0] * x + R[c * 3 + 1] * y + R[c * 3 + 2] * zc + T[c];
  }
  if (t < 144) {  // kv points: 4/head -> k_pts (row major), 8/head -> v_pts^T
    float x = kvpraw[n * 432 + t], y = kvpraw[n * 432 + 144 + t],
          zc = kvpraw[n * 432 + 288 + t];
    int h = t / 12, u = t % 12;
    float p[3];
#pragma unroll
    for (int c = 0; c < 3; ++c)
      p[c] = R[c * 3 + 0] * x + R[c * 3 + 1] * y + R[c * 3 + 2] * zc + T[c];
    if (u < 4) {
#pragma unroll
      for (int c = 0; c < 3; ++c) kpts[n * 144 + h * 12 + u * 3 + c] = p[c];
    } else {
#pragma unroll
      for (int c = 0; c < 3; ++c)
        vptsT[(size_t)(h * 24 + (u - 4) * 3 + c) * NRES + n] = p[c];
    }
  }
  __syncthreads();
  if (t < NHEAD) {
    float sq = 0.f, sk = 0.f;
#pragma unroll
    for (int j = 0; j < 12; ++j) {
      float aq = qpts[n * 144 + t * 12 + j];
      float ak = kpts[n * 144 + t * 12 + j];
      sq += aq * aq;
      sk += ak * ak;
    }
    qsq[n * NHEAD + t] = sq;
    ksq[n * NHEAD + t] = sk;
  }
}

// ---------------------------------------------------------------------------
// Attention logits + softmax. One wave per (q-tile, head).
// K lanes 0..15 = (q*sqrt(1/48)) . k ; K lanes 16..27 = (hw*qp) . kp ; 28..31=0.
// C preloaded with sqrt(1/3)*bias + mask - 0.5*hw*(qsq+ksq).
// Softmax: b128 LDS scans, exp computed once in place, packed bf16 b64 stores.
// ---------------------------------------------------------------------------
__global__ void __launch_bounds__(32)
attn_kernel(const float* __restrict__ qbuf, const float* __restrict__ kbuf,
            const float* __restrict__ qpts, const float* __restrict__ kpts,
            const float* __restrict__ qsq, const float* __restrict__ ksq,
            const float* __restrict__ biasbuf, const float* __restrict__ msk,
            const float* __restrict__ head_weights, __bf16* __restrict__ probs) {
  const int qt = blockIdx.x / NHEAD;
  const int h = blockIdx.x % NHEAD;
  const int lane = threadIdx.x & 31;
  const int nlo = lane & 15, half = lane >> 4;
  const int q0 = qt * 16;

  __shared__ float logits[16 * NRES];  // 48 KB of the WGP's 320 KB LDS
  __shared__ float red[64];

  const float s1 = 0.14433756729740643f;  // sqrt(1/48)
  const float s2 = 0.57735026918962576f;  // sqrt(1/3)
  float hwv = head_weights[h];
  hwv = log1pf(expf(hwv)) * 0.13608276348795434f;  // softplus * sqrt(1/54)

  // A operand (query side), invariant over k tiles: two contiguous runs.
  v16bf a;
  {
    const int qrow = q0 + nlo;
    float aq[8], ap[8];
    load8f(qbuf + qrow * 192 + h * 16 + half * 8, aq);
    load8f(qpts + qrow * 144 + h * 12 + half * 8, ap);  // padded buffer
#pragma unroll
    for (int e = 0; e < 8; ++e) {
      a[e] = (__bf16)(aq[e] * s1);
      int j = half * 8 + e;                 // point-dim index 0..15
      float f = (j < 12) ? ap[e] * hwv : 0.f;
      a[8 + e] = (__bf16)f;
    }
  }

  for (int kt = 0; kt < NRES / 16; ++kt) {
    const int kc = kt * 16 + nlo;
    // B operand: elements 0..7 carry k-features half*8.., 8..15 carry k-points half*8...
    const float* __restrict__ bb = kbuf + kc * 192 + h * 16 + half * 8;
    const float* __restrict__ bp = kpts + kc * 144 + h * 12 + half * 8;
    float bf[16];
    load8f(bb, bf);
    load8f(bp, bf + 8);
    v16bf b;
#pragma unroll
    for (int e = 0; e < 16; ++e) {
      float f = bf[e];
      if (e >= 12) f = half ? 0.f : f;      // point dims 12..15 are zero
      b[e] = (__bf16)f;
    }
    v8f c;
#pragma unroll
    for (int v = 0; v < 8; ++v) {
      int qq = q0 + v + half * 8;
      c[v] = s2 * biasbuf[((size_t)qq * NRES + kc) * NHEAD + h] +
             msk[(size_t)qq * NRES + kc] -
             0.5f * hwv * (qsq[qq * NHEAD + h] + ksq[kc * NHEAD + h]);
    }
    c = wmma_bf16(a, b, c);
#pragma unroll
    for (int v = 0; v < 8; ++v) logits[(v + half * 8) * NRES + kc] = c[v];
  }
  __syncthreads();

  // Row softmax: two lanes per row, each scans half the keys with b128 loads.
  const int row = nlo;
  float* __restrict__ lrow = logits + row * NRES + half * (NRES / 2);

  float mx = -1e30f;                        // pass 1: running max (v_max_num)
  for (int k = 0; k < NRES / 2; k += 4) {
    float4 u = *(const float4*)(lrow + k);
    mx = fmaxq(fmaxq(u.x, u.y), fmaxq(fmaxq(u.z, u.w), mx));
  }
  red[lane] = mx;
  __syncthreads();
  const float rmx = fmaxq(red[row], red[row + 16]);

  float sum = 0.f;                          // pass 2: exp in place + sum
  for (int k = 0; k < NRES / 2; k += 4) {
    float4 u = *(const float4*)(lrow + k);
    u.x = __expf(u.x - rmx);
    u.y = __expf(u.y - rmx);
    u.z = __expf(u.z - rmx);
    u.w = __expf(u.w - rmx);
    sum += (u.x + u.y) + (u.z + u.w);
    *(float4*)(lrow + k) = u;
  }
  red[32 + lane] = sum;
  __syncthreads();
  const float inv = 1.f / (red[32 + row] + red[32 + row + 16]);

  // pass 3: scale + pack 4 bf16 -> one b64 store.
  __bf16* __restrict__ prow =
      probs + ((size_t)h * NRES + q0 + row) * NRES + half * (NRES / 2);
  for (int k = 0; k < NRES / 2; k += 4) {
    float4 u = *(const float4*)(lrow + k);
    v4bf pk;
    pk[0] = (__bf16)(u.x * inv);
    pk[1] = (__bf16)(u.y * inv);
    pk[2] = (__bf16)(u.z * inv);
    pk[3] = (__bf16)(u.w * inv);
    *(v4bf*)(prow + k) = pk;
  }
}

// ---------------------------------------------------------------------------
// o = attn@v, o_pts = attn@v_pts.  One wave per (q-tile, head); B operands
// come from transposed V / V_pts so every fragment is a contiguous 16-run.
// ---------------------------------------------------------------------------
__global__ void __launch_bounds__(32)
av_kernel(const __bf16* __restrict__ probs, const float* __restrict__ vT,
          const float* __restrict__ vptsT, float* __restrict__ obuf,
          float* __restrict__ optsbuf) {
  const int qt = blockIdx.x / NHEAD;
  const int h = blockIdx.x % NHEAD;
  const int lane = threadIdx.x & 31;
  const int nlo = lane & 15, half = lane >> 4;
  const int q0 = qt * 16;

  const __bf16* __restrict__ arow = probs + ((size_t)h * NRES + q0 + nlo) * NRES;
  const float* __restrict__ bo_row = vT + (size_t)(h * 16 + nlo) * NRES;
  const float* __restrict__ bp0_row = vptsT + (size_t)(h * 24 + nlo) * NRES;
  const int j1 = 16 + (nlo & 7);  // clamped row for the 8-wide second tile
  const float* __restrict__ bp1_row = vptsT + (size_t)(h * 24 + j1) * NRES;
  const bool p1ok = (nlo < 8);

  v8f acc_o = {}, acc_p0 = {}, acc_p1 = {};
  for (int k0 = 0; k0 < NRES; k0 += 32) {
    v16bf a;
    {
      v8bf u0 = *(const v8bf*)(arow + k0 + half * 16);
      v8bf u1 = *(const v8bf*)(arow + k0 + half * 16 + 8);
#pragma unroll
      for (int e = 0; e < 8; ++e) { a[e] = u0[e]; a[8 + e] = u1[e]; }
    }
    float fo[16], f0[16], f1[16];
    load8f(bo_row + k0 + half * 16, fo);
    load8f(bo_row + k0 + half * 16 + 8, fo + 8);
    load8f(bp0_row + k0 + half * 16, f0);
    load8f(bp0_row + k0 + half * 16 + 8, f0 + 8);
    load8f(bp1_row + k0 + half * 16, f1);
    load8f(bp1_row + k0 + half * 16 + 8, f1 + 8);
    v16bf bo, bp0, bp1;
#pragma unroll
    for (int e = 0; e < 16; ++e) {
      bo[e] = (__bf16)fo[e];
      bp0[e] = (__bf16)f0[e];
      bp1[e] = (__bf16)(p1ok ? f1[e] : 0.f);
    }
    acc_o = wmma_bf16(a, bo, acc_o);
    acc_p0 = wmma_bf16(a, bp0, acc_p0);
    acc_p1 = wmma_bf16(a, bp1, acc_p1);
  }
#pragma unroll
  for (int v = 0; v < 8; ++v) {
    int qrow = q0 + v + half * 8;
    obuf[qrow * 192 + h * 16 + nlo] = acc_o[v];
    optsbuf[(qrow * NHEAD + h) * 24 + nlo] = acc_p0[v];
    if (p1ok) optsbuf[(qrow * NHEAD + h) * 24 + 16 + nlo] = acc_p1[v];
  }
}

// ---------------------------------------------------------------------------
// o_pair[q,h,c] = sum_k probs[h,q,k] * z[q,k,c].  One wave per query.
// M = heads (12->16), N = 128 (8 tiles), K = 768 (24 steps).  The 32x128 z
// block is staged through LDS with coalesced b128 loads (single HBM pass
// over z at the 23.3 TB/s roofline), B fragments then read from LDS.
// ---------------------------------------------------------------------------
__global__ void __launch_bounds__(32)
opair_kernel(const __bf16* __restrict__ probs, const float* __restrict__ z,
             float* __restrict__ opair) {
  const int q = blockIdx.x;
  const int lane = threadIdx.x & 31;
  const int nlo = lane & 15, half = lane >> 4;

  __shared__ float zt[32 * DPAIR];  // 16 KB

  const int mm = (nlo < NHEAD) ? nlo : (NHEAD - 1);
  const bool mok = (nlo < NHEAD);
  const __bf16* __restrict__ arow = probs + ((size_t)mm * NRES + q) * NRES;

  v8f acc[8] = {};
  for (int k0 = 0; k0 < NRES; k0 += 32) {
    // Cooperative, coalesced z stage: lane copies row (k0+lane), 32 x b128.
    const float* __restrict__ zr = z + ((size_t)q * NRES + k0 + lane) * DPAIR;
    float* __restrict__ dst = zt + lane * DPAIR;
#pragma unroll
    for (int i = 0; i < DPAIR / 4; ++i)
      ((float4*)dst)[i] = ((const float4*)zr)[i];
    if (k0 + 32 < NRES)  // hint next block toward L2 (global_prefetch_b8)
      __builtin_prefetch(z + ((size_t)q * NRES + k0 + 32 + lane) * DPAIR, 0, 1);
    __syncthreads();

    v16bf a;
    {
      v8bf u0 = *(const v8bf*)(arow + k0 + half * 8);
      v8bf u1 = *(const v8bf*)(arow + k0 + 16 + half * 8);
#pragma unroll
      for (int e = 0; e < 8; ++e) {
        a[e] = mok ? u0[e] : (__bf16)0.f;
        a[8 + e] = mok ? u1[e] : (__bf16)0.f;
      }
    }
#pragma unroll
    for (int nt = 0; nt < 8; ++nt) {
      v16bf b;
#pragma unroll
      for (int e = 0; e < 16; ++e)
        b[e] = (__bf16)zt[(half * 8 + (e & 7) + (e >> 3) * 16) * DPAIR + nt * 16 + nlo];
      acc[nt] = wmma_bf16(a, b, acc[nt]);
    }
    __syncthreads();
  }
#pragma unroll
  for (int nt = 0; nt < 8; ++nt)
#pragma unroll
    for (int v = 0; v < 8; ++v) {
      int m = v + half * 8;
      if (m < NHEAD)
        opair[((size_t)q * NHEAD + m) * DPAIR + nt * 16 + nlo] = acc[nt][v];
    }
}

// ---------------------------------------------------------------------------
// Finalize: inverse-frame output points, norms, concat cat[768,2112].
// ---------------------------------------------------------------------------
__global__ void __launch_bounds__(96)
finalize_kernel(const float* __restrict__ obuf, const float* __restrict__ optsbuf,
                const float* __restrict__ opair, const float* __restrict__ rot,
                const float* __restrict__ trans, float* __restrict__ cat) {
  const int n = blockIdx.x;
  const int t = threadIdx.x;  // 96 = H*PV
  float* crow = cat + (size_t)n * CATDIM;

  crow[t] = obuf[n * 192 + t];  // o   [0,192)
  crow[96 + t] = obuf[n * 192 + 96 + t];

  float R[9], T[3];
#pragma unroll
  for (int i = 0; i < 9; ++i) R[i] = rot[n * 9 + i];
#pragma unroll
  for (int i = 0; i < 3; ++i) T[i] = trans[n * 3 + i];

  {  // point t = h*8+p ; inverse frame: l_i = sum_j R[j][i]*(g_j - T_j)
    int h = t >> 3, p = t & 7;
    float g[3], l[3];
#pragma unroll
    for (int c = 0; c < 3; ++c)
      g[c] = optsbuf[(n * NHEAD + h) * 24 + p * 3 + c] - T[c];
#pragma unroll
    for (int i = 0; i < 3; ++i)
      l[i] = R[0 * 3 + i] * g[0] + R[1 * 3 + i] * g[1] + R[2 * 3 + i] * g[2];
    crow[192 + t] = l[0];  // x   [192,288)
    crow[288 + t] = l[1];  // y   [288,384)
    crow[384 + t] = l[2];  // z   [384,480)
    crow[480 + t] = sqrtf(l[0] * l[0] + l[1] * l[1] + l[2] * l[2] + 1e-8f);
  }
#pragma unroll
  for (int i = 0; i < 16; ++i)  // o_pair [576,2112)
    crow[576 + t * 16 + i] = opair[(size_t)n * (NHEAD * DPAIR) + t * 16 + i];
}

// ---------------------------------------------------------------------------
// Host-side launcher.
// ---------------------------------------------------------------------------
extern "C" void kernel_launch(void* const* d_in, const int* in_sizes, int n_in,
                              void* d_out, int out_size, void* d_ws, size_t ws_size,
                              hipStream_t stream) {
  const float* s = (const float*)d_in[0];
  const float* z = (const float*)d_in[1];
  const float* msk = (const float*)d_in[2];
  const float* rot = (const float*)d_in[3];
  const float* trans = (const float*)d_in[4];
  const float* Wq = (const float*)d_in[5];
  const float* bq = (const float*)d_in[6];
  const float* Wkv = (const float*)d_in[7];
  const float* bkv = (const float*)d_in[8];
  const float* Wqp = (const float*)d_in[9];
  const float* bqp = (const float*)d_in[10];
  const float* Wkvp = (const float*)d_in[11];
  const float* bkvp = (const float*)d_in[12];
  const float* Wb = (const float*)d_in[13];
  const float* bb = (const float*)d_in[14];
  const float* hwt = (const float*)d_in[15];
  const float* Wout = (const float*)d_in[16];
  const float* bout = (const float*)d_in[17];
  float* out = (float*)d_out;

  float* ws = (float*)d_ws;
  size_t o = 0;
  const size_t OFF_Q = o;      o += (size_t)NRES * 192;
  const size_t OFF_KV = o;     o += (size_t)NRES * 384;
  const size_t OFF_QPR = o;    o += (size_t)NRES * 144;
  const size_t OFF_KVPR = o;   o += (size_t)NRES * 432;
  const size_t OFF_K = o;      o += (size_t)NRES * 192;
  const size_t OFF_QPT = o;    o += (size_t)NRES * 144 + 16;  // +pad for 16-wide
  const size_t OFF_KPT = o;    o += (size_t)NRES * 144 + 16;  //  point loads
  const size_t OFF_VT = o;     o += (size_t)192 * NRES;
  const size_t OFF_VPTT = o;   o += (size_t)288 * NRES;
  const size_t OFF_QSQ = o;    o += (size_t)NRES * NHEAD;
  const size_t OFF_KSQ = o;    o += (size_t)NRES * NHEAD;
  const size_t OFF_BIAS = o;   o += (size_t)NRES * NRES * NHEAD;
  const size_t OFF_O = o;      o += (size_t)NRES * 192;
  const size_t OFF_OPTS = o;   o += (size_t)NRES * NHEAD * 24;
  const size_t OFF_OPAIR = o;  o += (size_t)NRES * NHEAD * DPAIR;
  const size_t OFF_CAT = o;    o += (size_t)NRES * CATDIM;
  const size_t OFF_WQT = o;    o += (size_t)DSING * 192;
  const size_t OFF_WKVT = o;   o += (size_t)DSING * 384;
  const size_t OFF_WQPT = o;   o += (size_t)DSING * 144;
  const size_t OFF_WKVPT = o;  o += (size_t)DSING * 432;
  const size_t OFF_WBT = o;    o += (size_t)DPAIR * NHEAD;
  const size_t OFF_WOUTT = o;  o += (size_t)CATDIM * DSING;
  __bf16* probs = (__bf16*)((char*)d_ws + o * sizeof(float));

  // 0. One-off weight transposes.
  auto T = [&](const float* Bsrc, float* Bdst, int K, int N) {
    transpose_kernel<<<dim3((K * N + 255) / 256), 256, 0, stream>>>(Bsrc, Bdst, K, N);
  };
  T(Wq, ws + OFF_WQT, DSING, 192);
  T(Wkv, ws + OFF_WKVT, DSING, 384);
  T(Wqp, ws + OFF_WQPT, DSING, 144);
  T(Wkvp, ws + OFF_WKVPT, DSING, 432);
  T(Wb, ws + OFF_WBT, DPAIR, NHEAD);
  T(Wout, ws + OFF_WOUTT, CATDIM, DSING);

  // 1. Projections.
  gemm_bt<false><<<dim3(48 * 12), 32, 0, stream>>>(s, ws + OFF_WQT, bq, ws + OFF_Q,
                                                   NRES, 192, DSING, DSING, 192);
  gemm_bt<false><<<dim3(48 * 24), 32, 0, stream>>>(s, ws + OFF_WKVT, bkv, ws + OFF_KV,
                                                   NRES, 384, DSING, DSING, 384);
  gemm_bt<false><<<dim3(48 * 9), 32, 0, stream>>>(s, ws + OFF_WQPT, bqp, ws + OFF_QPR,
                                                  NRES, 144, DSING, DSING, 144);
  gemm_bt<false><<<dim3(48 * 27), 32, 0, stream>>>(s, ws + OFF_WKVPT, bkvp,
                                                   ws + OFF_KVPR, NRES, 432, DSING,
                                                   DSING, 432);
  // 2. bias = z @ Wb + bb  (z read #1; flattened 589824 x 128, N=12 guarded).
  gemm_bt<true><<<dim3((NRES * NRES) / 16), 32, 0, stream>>>(
      z, ws + OFF_WBT, bb, ws + OFF_BIAS, NRES * NRES, NHEAD, DPAIR, DPAIR, NHEAD);
  // 3. Frames.
  frames_kernel<<<dim3(NRES), 192, 0, stream>>>(
      ws + OFF_KV, ws + OFF_QPR, ws + OFF_KVPR, rot, trans, ws + OFF_K, ws + OFF_VT,
      ws + OFF_QPT, ws + OFF_KPT, ws + OFF_VPTT, ws + OFF_QSQ, ws + OFF_KSQ);
  // 4. Fused logits + softmax -> bf16 probs.
  attn_kernel<<<dim3(48 * NHEAD), 32, 0, stream>>>(
      ws + OFF_Q, ws + OFF_K, ws + OFF_QPT, ws + OFF_KPT, ws + OFF_QSQ,
      ws + OFF_KSQ, ws + OFF_BIAS, msk, hwt, probs);
  // 5. attn @ v, attn @ v_pts.
  av_kernel<<<dim3(48 * NHEAD), 32, 0, stream>>>(probs, ws + OFF_VT, ws + OFF_VPTT,
                                                 ws + OFF_O, ws + OFF_OPTS);
  // 6. o_pair (z read #2, LDS-staged).
  opair_kernel<<<dim3(NRES), 32, 0, stream>>>(probs, z, ws + OFF_OPAIR);
  // 7. Finalize / concat.
  finalize_kernel<<<dim3(NRES), 96, 0, stream>>>(ws + OFF_O, ws + OFF_OPTS,
                                                 ws + OFF_OPAIR, rot, trans,
                                                 ws + OFF_CAT);
  // 8. out = cat @ Wout + bout.
  gemm_bt<false><<<dim3(48 * 24), 32, 0, stream>>>(ws + OFF_CAT, ws + OFF_WOUTT,
                                                   bout, out, NRES, DSING, CATDIM,
                                                   CATDIM, DSING);
}
